// Retriever_52192442581252
// MI455X (gfx1250) — matrix operand order repaired and checked by reading, not compiled, not run on hardware
//
#include <hip/hip_runtime.h>
#include <hip/hip_bf16.h>
#include <stdint.h>

// ---------------------------------------------------------------------------
// CDNA5 (gfx1250) WMMA types
// ---------------------------------------------------------------------------
typedef __bf16 v16bf __attribute__((ext_vector_type(16)));
typedef float  v8f   __attribute__((ext_vector_type(8)));

union V16 {
    struct { uint4 lo, hi; } u;   // 2 x 16B = 32B
    v16bf v;
};

#define EMBD 64
#define HE_K 96          // 74 real + 22 zero pad, multiple of 32
#define EDGE_K 128       // [q_emb | edge_attr]
#define ASD_N 128        // [A_s | A_d]

// ---------------------------------------------------------------------------
// Small utility kernels
// ---------------------------------------------------------------------------
__global__ void zero_f_kernel(float* p, int n) {
    int i = blockIdx.x * blockDim.x + threadIdx.x;
    if (i < n) p[i] = 0.0f;
}

__global__ void out_init_kernel(float* out, const float* b2, int E) {
    int e = blockIdx.x * blockDim.x + threadIdx.x;
    if (e < E) out[e] = b2[0];
}

// in-degree (dst) and out-degree (src) as floats
__global__ void degrees_kernel(const long long* ei, float* deg_dst,
                               float* deg_src, int E) {
    int e = blockIdx.x * blockDim.x + threadIdx.x;
    if (e >= E) return;
    atomicAdd(&deg_dst[(int)ei[(size_t)E + e]], 1.0f);
    atomicAdd(&deg_src[(int)ei[e]], 1.0f);
}

// one DDE round: acc[to] += h[from]  (2 components)
__global__ void scatter_kernel(const long long* ei, const float* hin,
                               float* acc, int E, int reverse) {
    int e = blockIdx.x * blockDim.x + threadIdx.x;
    if (e >= E) return;
    int s = (int)ei[e];
    int d = (int)ei[(size_t)E + e];
    int from = reverse ? d : s;
    int to   = reverse ? s : d;
    atomicAdd(&acc[2 * to + 0], hin[2 * from + 0]);
    atomicAdd(&acc[2 * to + 1], hin[2 * from + 1]);
}

__global__ void normalize_kernel(const float* acc, const float* deg,
                                 float* hout, int N) {
    int i = blockIdx.x * blockDim.x + threadIdx.x;
    if (i >= 2 * N) return;
    hout[i] = acc[i] / fmaxf(deg[i >> 1], 1.0f);
}

// ---------------------------------------------------------------------------
// h_e build: [N, 96] bf16  = [ x_fixed(64) | topic(2) | rounds(8) | pad(22) ]
// ---------------------------------------------------------------------------
__global__ void build_he_kernel(const float* x, const float* topic,
                                const float* rounds, const float* non_text,
                                __bf16* he, int N) {
    int n = blockIdx.x * blockDim.x + threadIdx.x;
    if (n >= N) return;
    const float* xr = x + (size_t)n * EMBD;
    float vals[EMBD];
    bool allzero = true;
    for (int j = 0; j < EMBD; j += 4) {
        float4 f = *reinterpret_cast<const float4*>(xr + j);
        vals[j + 0] = f.x; vals[j + 1] = f.y;
        vals[j + 2] = f.z; vals[j + 3] = f.w;
        allzero = allzero & (f.x == 0.0f) & (f.y == 0.0f) &
                            (f.z == 0.0f) & (f.w == 0.0f);
    }
    __bf16* o = he + (size_t)n * HE_K;
    for (int j = 0; j < EMBD; ++j)
        o[j] = (__bf16)(allzero ? non_text[j] : vals[j]);
    o[64] = (__bf16)topic[2 * n + 0];
    o[65] = (__bf16)topic[2 * n + 1];
    for (int r = 0; r < 4; ++r) {
        o[66 + 2 * r + 0] = (__bf16)rounds[(size_t)r * N * 2 + 2 * n + 0];
        o[66 + 2 * r + 1] = (__bf16)rounds[(size_t)r * N * 2 + 2 * n + 1];
    }
    for (int j = 74; j < HE_K; ++j) o[j] = (__bf16)0.0f;
}

// ---------------------------------------------------------------------------
// Weight repack (transposed, bf16, K-contiguous rows for B-operand b128 loads)
// WqaT[n][k] (64 x 128): k<64 -> W1[k][n], else W1[138 + k-64][n]
// WsdT[n][k] (128 x 96): n<64 -> W1[64+k][n] ; n>=64 -> W1[202+k][n-64]; k>=74 pad
// ---------------------------------------------------------------------------
__global__ void prep_wqaT_kernel(const float* W1, __bf16* WqaT) {
    int i = blockIdx.x * blockDim.x + threadIdx.x;
    if (i >= EMBD * EDGE_K) return;
    int n = i >> 7, k = i & 127;
    int row = (k < 64) ? k : (138 + (k - 64));
    WqaT[i] = (__bf16)W1[row * EMBD + n];
}

__global__ void prep_wsdT_kernel(const float* W1, __bf16* WsdT) {
    int i = blockIdx.x * blockDim.x + threadIdx.x;
    if (i >= ASD_N * HE_K) return;
    int n = i / HE_K, k = i % HE_K;
    float v = 0.0f;
    if (k < 74) {
        int col = (n < 64) ? n : (n - 64);
        int row = ((n < 64) ? 64 : 202) + k;
        v = W1[row * EMBD + col];
    }
    WsdT[i] = (__bf16)v;
}

// ---------------------------------------------------------------------------
// Node GEMM: A_sd[N,128] = he[N,96](bf16) @ WsdT^T   (f32 accumulate)
// 4 waves per block; each wave owns one 16x16 tile; K loop = 3 steps of 32.
// Tile grid is exact (N % 16 == 0) -> unguarded straight-line stores.
// ---------------------------------------------------------------------------
__global__ __launch_bounds__(128) void node_gemm_kernel(
    const __bf16* __restrict__ he, const __bf16* __restrict__ WsdT,
    float* __restrict__ A_sd, int N) {
    int tile = blockIdx.x * 4 + (threadIdx.x >> 5);
    int tm = tile >> 3;          // 8 N-tiles of 16 over ASD_N=128
    int tn = tile & 7;
    if (tm * 16 >= N) return;    // uniform per-wave guard only

    int lane = threadIdx.x & 31;
    int g = lane >> 4;           // lane group
    int nloc = lane & 15;
    int arow = tm * 16 + nloc;   // A-operand: both lane groups hold M=0..15

    v8f acc = {};
    for (int ks = 0; ks < HE_K / 32; ++ks) {
        V16 a;
        const __bf16* ap = he + (size_t)arow * HE_K + ks * 32 + g * 8;
        a.u.lo = *reinterpret_cast<const uint4*>(ap);        // K: base .. base+7
        a.u.hi = *reinterpret_cast<const uint4*>(ap + 16);   // K: base+16 .. +23
        V16 b;
        const __bf16* bp = WsdT + (size_t)(tn * 16 + nloc) * HE_K + ks * 32 + g * 8;
        b.u.lo = *reinterpret_cast<const uint4*>(bp);        // K: base .. base+7
        b.u.hi = *reinterpret_cast<const uint4*>(bp + 16);   // K: base+16 .. +23
        acc = __builtin_amdgcn_wmma_f32_16x16x32_bf16(
            false, a.v, false, b.v, (short)0, acc, false, false);
    }
    // D layout: VGPR v -> M = v + 8*g, N = nloc
    float* orow = A_sd + (size_t)(tm * 16 + 8 * g) * ASD_N + tn * 16 + nloc;
    #pragma unroll
    for (int v = 0; v < 8; ++v)
        orow[(size_t)v * ASD_N] = acc[v];
}

// ---------------------------------------------------------------------------
// Edge kernel: 16 edges per block, 4 waves (one per 16-col slice of EMB=64).
//  hidden = relu( [q|ea] @ WqaT^T + A_sd[src][0:64] + A_sd[dst][64:128] + b1 )
//  out[e] += hidden . W2   (out pre-initialized to b2)
// ---------------------------------------------------------------------------
__global__ __launch_bounds__(128) void edge_kernel(
    const float* __restrict__ q_emb, const float* __restrict__ edge_attr,
    const long long* __restrict__ ei, const float* __restrict__ A_sd,
    const float* __restrict__ b1, const float* __restrict__ W2,
    const __bf16* __restrict__ WqaT, float* __restrict__ out, int E) {

    __shared__ __bf16 Asm[16 * EDGE_K];   // 4 KB staged A tile
    __shared__ int srcs[16], dsts[16];

    int t = threadIdx.x;
    int e0 = blockIdx.x * 16;
    bool full = (e0 + 16 <= E);           // uniform per block

    if (t < 16) {
        int e = e0 + t;
        if (e >= E) e = E - 1;
        int s = (int)ei[e];
        int d = (int)ei[(size_t)E + e];
        srcs[t] = s;
        dsts[t] = d;
        // warm L2 for the gathered epilogue rows while the WMMAs run
        __builtin_prefetch(A_sd + (size_t)s * ASD_N, 0, 3);
        __builtin_prefetch(A_sd + (size_t)d * ASD_N + EMBD, 0, 3);
    }
    // Stage 16x128 bf16 A tile: thread t loads 16 floats (row t>>3, chunk t&7)
    {
        int r = t >> 3;
        int c = (t & 7) * 16;
        int e = e0 + r;
        if (e >= E) e = E - 1;
        const float* srow = (c < EMBD)
            ? (q_emb     + (size_t)e * EMBD + c)
            : (edge_attr + (size_t)e * EMBD + (c - EMBD));
        __bf16* drow = &Asm[r * EDGE_K + c];
        for (int j = 0; j < 16; j += 4) {
            float4 f = *reinterpret_cast<const float4*>(srow + j);
            drow[j + 0] = (__bf16)f.x; drow[j + 1] = (__bf16)f.y;
            drow[j + 2] = (__bf16)f.z; drow[j + 3] = (__bf16)f.w;
        }
    }
    __syncthreads();

    int wave = t >> 5;
    int lane = t & 31;
    int g = lane >> 4;
    int nloc = lane & 15;
    int nglob = wave * 16 + nloc;

    v8f acc = {};
    for (int ks = 0; ks < EDGE_K / 32; ++ks) {
        V16 a;
        const __bf16* ap = &Asm[nloc * EDGE_K + ks * 32 + g * 8];
        a.u.lo = *reinterpret_cast<const uint4*>(ap);
        a.u.hi = *reinterpret_cast<const uint4*>(ap + 16);
        V16 b;
        const __bf16* bp = WqaT + (size_t)nglob * EDGE_K + ks * 32 + g * 8;
        b.u.lo = *reinterpret_cast<const uint4*>(bp);
        b.u.hi = *reinterpret_cast<const uint4*>(bp + 16);
        acc = __builtin_amdgcn_wmma_f32_16x16x32_bf16(
            false, a.v, false, b.v, (short)0, acc, false, false);
    }

    float w2n = W2[nglob];
    float b1n = b1[nglob];
    #pragma unroll
    for (int v = 0; v < 8; ++v) {
        int m = v + 8 * g;                 // edge row within tile
        float h = acc[v]
                + A_sd[(size_t)srcs[m] * ASD_N + nglob]
                + A_sd[(size_t)dsts[m] * ASD_N + EMBD + nglob]
                + b1n;
        h = fmaxf(h, 0.0f);
        float s = h * w2n;
        // reduce over the 16 lanes holding this row's 16 columns
        s += __shfl_xor(s, 1, 32);
        s += __shfl_xor(s, 2, 32);
        s += __shfl_xor(s, 4, 32);
        s += __shfl_xor(s, 8, 32);
        if (nloc == 0 && (full || (e0 + m) < E))
            atomicAdd(&out[e0 + m], s);
    }
}

// ---------------------------------------------------------------------------
// Launch
// ---------------------------------------------------------------------------
extern "C" void kernel_launch(void* const* d_in, const int* in_sizes, int n_in,
                              void* d_out, int out_size, void* d_ws, size_t ws_size,
                              hipStream_t stream) {
    const float*      x         = (const float*)d_in[0];
    const long long*  ei        = (const long long*)d_in[1];   // int64 in reference
    const float*      edge_attr = (const float*)d_in[2];
    const float*      topic     = (const float*)d_in[3];
    const float*      q_emb     = (const float*)d_in[4];
    const float*      non_text  = (const float*)d_in[5];
    const float*      W1        = (const float*)d_in[6];
    const float*      b1        = (const float*)d_in[7];
    const float*      W2        = (const float*)d_in[8];
    const float*      b2        = (const float*)d_in[9];
    float*            out       = (float*)d_out;

    const int N = in_sizes[0] / EMBD;
    const int E = in_sizes[2] / EMBD;

    // ---- workspace carve (bump allocator, 256B aligned) ----
    uintptr_t base = (uintptr_t)d_ws;
    auto alloc = [&](size_t bytes) -> void* {
        void* p = (void*)base;
        base += (bytes + 255) & ~(size_t)255;
        return p;
    };
    __bf16* he     = (__bf16*)alloc((size_t)N * HE_K * sizeof(__bf16));
    __bf16* WqaT   = (__bf16*)alloc((size_t)EMBD * EDGE_K * sizeof(__bf16));
    __bf16* WsdT   = (__bf16*)alloc((size_t)ASD_N * HE_K * sizeof(__bf16));
    float*  A_sd   = (float*) alloc((size_t)N * ASD_N * sizeof(float));
    float*  degd   = (float*) alloc((size_t)N * sizeof(float));
    float*  degs   = (float*) alloc((size_t)N * sizeof(float));
    float*  acc    = (float*) alloc((size_t)N * 2 * sizeof(float));
    float*  rounds = (float*) alloc((size_t)4 * N * 2 * sizeof(float));

    const int TB = 256;
    auto g1 = [&](int n) { return dim3((n + TB - 1) / TB); };

    // degrees + out init
    zero_f_kernel<<<g1(N), TB, 0, stream>>>(degd, N);
    zero_f_kernel<<<g1(N), TB, 0, stream>>>(degs, N);
    out_init_kernel<<<g1(E), TB, 0, stream>>>(out, b2, E);
    degrees_kernel<<<g1(E), TB, 0, stream>>>(ei, degd, degs, E);

    // DDE rounds: r0,r1 forward (seg=dst, deg=degd); r2,r3 reverse (seg=src)
    for (int r = 0; r < 4; ++r) {
        int reverse = (r >= 2);
        const float* hin = (r == 0 || r == 2) ? topic
                                              : (rounds + (size_t)(r - 1) * N * 2);
        const float* deg = reverse ? degs : degd;
        float* hout = rounds + (size_t)r * N * 2;
        zero_f_kernel<<<g1(2 * N), TB, 0, stream>>>(acc, 2 * N);
        scatter_kernel<<<g1(E), TB, 0, stream>>>(ei, hin, acc, E, reverse);
        normalize_kernel<<<g1(2 * N), TB, 0, stream>>>(acc, deg, hout, N);
    }

    // node features + weight repack
    build_he_kernel<<<g1(N), 128, 0, stream>>>(x, topic, rounds, non_text, he, N);
    prep_wqaT_kernel<<<g1(EMBD * EDGE_K), TB, 0, stream>>>(W1, WqaT);
    prep_wsdT_kernel<<<g1(ASD_N * HE_K), TB, 0, stream>>>(W1, WsdT);

    // node GEMM: A_sd = he @ [W1_s | W1_d]   (4 wave-tiles per block)
    int n_tiles = ((N + 15) / 16) * (ASD_N / 16);
    node_gemm_kernel<<<dim3((n_tiles + 3) / 4), 128, 0, stream>>>(he, WsdT, A_sd, N);

    // edge GEMM + epilogue (relu, @W2, +b2 via pre-init)
    int e_blocks = (E + 15) / 16;
    edge_kernel<<<dim3(e_blocks), 128, 0, stream>>>(
        q_emb, edge_attr, ei, A_sd, b1, W2, WqaT, out, E);
}
